// GNN_80599356277028
// MI455X (gfx1250) — matrix operand order repaired and checked by reading, not compiled, not run on hardware
//
#include <hip/hip_runtime.h>
#include <hip/hip_bf16.h>

typedef _Float16 half8  __attribute__((ext_vector_type(8)));
typedef _Float16 half16 __attribute__((ext_vector_type(16)));
typedef float    v8f    __attribute__((ext_vector_type(8)));

#define F 128          // feature width (IN_CH == HID == 128)
#define KPAD_C 288     // 2*HID + EDGE_FEAT = 259 padded to multiple of 32
#define LDSK_N 136     // LDS row stride (f16) for node weights: 68 dw = 4 mod 64 -> conflict-free
#define LDSK_C 296     // LDS row stride (f16) for edge weights: 148 dw = 20 mod 64 -> conflict-free

// ---- WMMA fragment loaders (layouts per CDNA5 ISA §7.12.2) ----
// A 16x32 f16: lane L -> M = L&15 ; K = kb..kb+7 and kb+16..kb+23, kb=(L>>4)*8
__device__ __forceinline__ half16 load_a_frag(const _Float16* rowPtr, int lane) {
    int kb = (lane >> 4) * 8;
    half8 lo = *(const half8*)(rowPtr + kb);
    half8 hi = *(const half8*)(rowPtr + kb + 16);
    half16 r;
#pragma unroll
    for (int i = 0; i < 8; ++i) { r[i] = lo[i]; r[i + 8] = hi[i]; }
    return r;
}
// B 32x16 f16 (from W^T, n-major rows): lane L -> N = L&15 ;
// K = kk*32 + (L>>4)*8 + 0..7 and +16..23 (matches A). rowPtr may be LDS or global.
__device__ __forceinline__ half16 load_b_frag(const _Float16* rowPtr, int kk, int lane) {
    int k0 = kk * 32 + ((lane >> 4) << 3);
    half8 lo = *(const half8*)(rowPtr + k0);
    half8 hi = *(const half8*)(rowPtr + k0 + 16);
    half16 r;
#pragma unroll
    for (int i = 0; i < 8; ++i) { r[i] = lo[i]; r[i + 8] = hi[i]; }
    return r;
}

// ---- small utility kernels ----
__global__ void gnn_w_transpose_f16(const float* __restrict__ W, _Float16* __restrict__ WT,
                                    int K, int Ncols, int Kpad) {
    int i = blockIdx.x * blockDim.x + threadIdx.x;
    if (i >= Ncols * Kpad) return;
    int n = i / Kpad, k = i % Kpad;
    float v = (k < K) ? W[(size_t)k * Ncols + n] : 0.0f;
    WT[(size_t)n * Kpad + k] = (_Float16)v;
}

__global__ void gnn_f32_to_f16(const float* __restrict__ in, _Float16* __restrict__ out, int n) {
    int i = blockIdx.x * blockDim.x + threadIdx.x;
    if (i < n) out[i] = (_Float16)in[i];
}

__global__ void gnn_mean_to_f16(const float* __restrict__ agg, const float* __restrict__ cnt,
                                _Float16* __restrict__ out, int total) {
    int i = blockIdx.x * blockDim.x + threadIdx.x;
    if (i >= total) return;
    float c = fmaxf(cnt[i >> 7], 1.0f);
    out[i] = (_Float16)(agg[i] / c);
}

// 32 lanes per edge, float4 per lane, f32 atomics into L2-resident accumulator
__global__ void gnn_scatter_add(const float* __restrict__ feat, const int* __restrict__ src,
                                const int* __restrict__ dst, float* __restrict__ agg,
                                float* __restrict__ cnt, int E) {
    int t = blockIdx.x * blockDim.x + threadIdx.x;
    int e = t >> 5;
    if (e >= E) return;
    int lane = t & 31;
    int s = src[e], d = dst[e];
    float4 v = ((const float4*)(feat + (size_t)s * F))[lane];
    float* p = agg + (size_t)d * F + lane * 4;
    atomicAdd(p + 0, v.x);
    atomicAdd(p + 1, v.y);
    atomicAdd(p + 2, v.z);
    atomicAdd(p + 3, v.w);
    if (cnt && lane == 0) atomicAdd(&cnt[d], 1.0f);
}

// ---- fused SAGE node GEMM: out = act(A1@B1 + A2@B2 + bias), wave = 16x128 tile ----
// Weights staged in LDS once per block (saves ~200 MB of redundant L2 reads).
__global__ void gnn_node_gemm(const _Float16* __restrict__ A1, const _Float16* __restrict__ A2,
                              const _Float16* __restrict__ B1T, const _Float16* __restrict__ B2T,
                              const float* __restrict__ bias, float* __restrict__ outF,
                              _Float16* __restrict__ outH, int nNodes, int relu) {
    __shared__ _Float16 sB[2 * F * LDSK_N];   // 69632 bytes (< 320KB/WGP)

    // cooperative weight staging: 2 matrices x 128 rows x 16 chunks of 8 f16
    for (int c = threadIdx.x; c < 2 * F * (F / 8); c += blockDim.x) {
        int mat = c >> 11;                      // / 2048
        int cc = c & 2047;
        int n = cc >> 4;                        // 16 chunks per row
        int k8 = (cc & 15) * 8;
        const _Float16* sp = (mat ? B2T : B1T) + (size_t)n * F + k8;
        *(half8*)&sB[mat * F * LDSK_N + n * LDSK_N + k8] = *(const half8*)sp;
    }
    __syncthreads();

    int wave = blockIdx.x * (blockDim.x >> 5) + (threadIdx.x >> 5);
    int m0 = wave * 16;
    if (m0 >= nNodes) return;                  // wave-uniform, after barrier
    int lane = threadIdx.x & 31;
    int mr = lane & 15;
    int mrow = min(m0 + mr, nNodes - 1);
    const _Float16* r1 = A1 + (size_t)mrow * F;
    const _Float16* r2 = A2 + (size_t)mrow * F;

    v8f acc[8];
#pragma unroll
    for (int nt = 0; nt < 8; ++nt)
#pragma unroll
        for (int r = 0; r < 8; ++r) acc[nt][r] = 0.0f;

#pragma unroll
    for (int kk = 0; kk < 4; ++kk) {
        half16 a1 = load_a_frag(r1 + kk * 32, lane);
        half16 a2 = load_a_frag(r2 + kk * 32, lane);
#pragma unroll
        for (int nt = 0; nt < 8; ++nt) {
            const _Float16* b1row = &sB[(nt * 16 + mr) * LDSK_N];
            const _Float16* b2row = &sB[F * LDSK_N + (nt * 16 + mr) * LDSK_N];
            half16 b1 = load_b_frag(b1row, kk, lane);
            half16 b2 = load_b_frag(b2row, kk, lane);
            acc[nt] = __builtin_amdgcn_wmma_f32_16x16x32_f16(false, a1, false, b1,
                                                             (short)0, acc[nt], false, false);
            acc[nt] = __builtin_amdgcn_wmma_f32_16x16x32_f16(false, a2, false, b2,
                                                             (short)0, acc[nt], false, false);
        }
    }

    int hi = lane >> 4;
#pragma unroll
    for (int nt = 0; nt < 8; ++nt) {
        int n = nt * 16 + mr;
        float bv = bias[n];
#pragma unroll
        for (int r = 0; r < 8; ++r) {
            int m = m0 + r + 8 * hi;
            if (m < nNodes) {
                float v = acc[nt][r] + bv;
                if (relu) v = fmaxf(v, 0.0f);
                outH[(size_t)m * F + n] = (_Float16)v;
                if (outF) outF[(size_t)m * F + n] = v;
            }
        }
    }
}

// ---- fully fused edge classifier: wave = 16 edges ----
// hidden(16x128) = ReLU(concat(h[src],h[dst],ea) @ Wc1 + bc1) kept in regs,
// out = hidden @ Wc2 + bc2 via shfl reduction. Wc1^T staged in LDS per block
// (cuts weight traffic from ~3.7 GB of L2 reads to ~460 MB).
__global__ void gnn_edge_mlp(const _Float16* __restrict__ h, const int* __restrict__ src,
                             const int* __restrict__ dst, const float* __restrict__ ea,
                             const _Float16* __restrict__ Wc1T, const float* __restrict__ bc1,
                             const float* __restrict__ Wc2, const float* __restrict__ bc2,
                             float* __restrict__ out, int E) {
    __shared__ _Float16 sW[F * LDSK_C];        // 75776 bytes (< 320KB/WGP)

    // cooperative weight staging: 128 rows x 36 chunks of 8 f16
    for (int c = threadIdx.x; c < F * (KPAD_C / 8); c += blockDim.x) {
        int n = c / (KPAD_C / 8);
        int k8 = (c % (KPAD_C / 8)) * 8;
        *(half8*)&sW[n * LDSK_C + k8] = *(const half8*)(Wc1T + (size_t)n * KPAD_C + k8);
    }
    __syncthreads();

    int tile = blockIdx.x * (blockDim.x >> 5) + (threadIdx.x >> 5);
    int eBase = tile * 16;
    if (eBase >= E) return;                    // wave-uniform, after barrier
    int lane = threadIdx.x & 31;
    int col = lane & 15;
    int e = min(eBase + col, E - 1);
    int s = src[e], d = dst[e];
    const _Float16* rs = h + (size_t)s * F;
    const _Float16* rd = h + (size_t)d * F;

    v8f acc[8];
#pragma unroll
    for (int nt = 0; nt < 8; ++nt)
#pragma unroll
        for (int r = 0; r < 8; ++r) acc[nt][r] = 0.0f;

#pragma unroll
    for (int kk = 0; kk < 9; ++kk) {
        half16 a;
        if (kk < 4) {
            a = load_a_frag(rs + kk * 32, lane);
        } else if (kk < 8) {
            a = load_a_frag(rd + (kk - 4) * 32, lane);
        } else {
            // K = 256..287: edge features at 256..258 (lo-half lanes), zero pad above
#pragma unroll
            for (int i = 0; i < 16; ++i) a[i] = (_Float16)0.0f;
            if (lane < 16) {
                a[0] = (_Float16)ea[e * 3 + 0];
                a[1] = (_Float16)ea[e * 3 + 1];
                a[2] = (_Float16)ea[e * 3 + 2];
            }
        }
#pragma unroll
        for (int nt = 0; nt < 8; ++nt) {
            const _Float16* brow = &sW[(nt * 16 + col) * LDSK_C];
            half16 b = load_b_frag(brow, kk, lane);
            acc[nt] = __builtin_amdgcn_wmma_f32_16x16x32_f16(false, a, false, b,
                                                             (short)0, acc[nt], false, false);
        }
    }

    // bias + ReLU + dot with Wc2, reduced over the 128 columns
    float partial[8];
#pragma unroll
    for (int r = 0; r < 8; ++r) partial[r] = 0.0f;
#pragma unroll
    for (int nt = 0; nt < 8; ++nt) {
        int n = nt * 16 + col;
        float bv = bc1[n];
        float w2 = Wc2[n];
#pragma unroll
        for (int r = 0; r < 8; ++r) {
            float hv = fmaxf(acc[nt][r] + bv, 0.0f);
            partial[r] += hv * w2;
        }
    }
#pragma unroll
    for (int r = 0; r < 8; ++r) {
#pragma unroll
        for (int off = 1; off < 16; off <<= 1)
            partial[r] += __shfl_xor(partial[r], off, 16);
    }
    if (col == 0) {
        float b2 = bc2[0];
        int mhi = (lane >> 4) * 8;
#pragma unroll
        for (int r = 0; r < 8; ++r) {
            int m = eBase + mhi + r;
            if (m < E) out[m] = partial[r] + b2;
        }
    }
}

extern "C" void kernel_launch(void* const* d_in, const int* in_sizes, int n_in,
                              void* d_out, int out_size, void* d_ws, size_t ws_size,
                              hipStream_t stream) {
    const float* x   = (const float*)d_in[0];
    const int*   ei  = (const int*)d_in[1];
    const float* ea  = (const float*)d_in[2];
    const float* W1l = (const float*)d_in[3];
    const float* b1  = (const float*)d_in[4];
    const float* W1r = (const float*)d_in[5];
    const float* W2l = (const float*)d_in[6];
    const float* b2  = (const float*)d_in[7];
    const float* W2r = (const float*)d_in[8];
    const float* Wc1 = (const float*)d_in[9];
    const float* bc1 = (const float*)d_in[10];
    const float* Wc2 = (const float*)d_in[11];
    const float* bc2 = (const float*)d_in[12];
    (void)n_in; (void)out_size; (void)ws_size;

    int N = in_sizes[0] / F;       // 50000
    int E = in_sizes[1] / 2;       // 800000
    const int* src = ei;
    const int* dstI = ei + E;
    float* out = (float*)d_out;

    // workspace carve-out
    char* base = (char*)d_ws;
    size_t off = 0;
    auto alloc = [&](size_t bytes) -> void* {
        void* p = base + off;
        off = (off + bytes + 255) & ~(size_t)255;
        return p;
    };
    float*     agg   = (float*)alloc((size_t)N * F * sizeof(float));
    float*     cnt   = (float*)alloc((size_t)N * sizeof(float));
    float*     h1f   = (float*)alloc((size_t)N * F * sizeof(float));
    _Float16*  xh    = (_Float16*)alloc((size_t)N * F * sizeof(_Float16));
    _Float16*  h1h   = (_Float16*)alloc((size_t)N * F * sizeof(_Float16));
    _Float16*  h2h   = (_Float16*)alloc((size_t)N * F * sizeof(_Float16));
    _Float16*  meanh = (_Float16*)alloc((size_t)N * F * sizeof(_Float16));
    _Float16*  W1lT  = (_Float16*)alloc((size_t)F * F * sizeof(_Float16));
    _Float16*  W1rT  = (_Float16*)alloc((size_t)F * F * sizeof(_Float16));
    _Float16*  W2lT  = (_Float16*)alloc((size_t)F * F * sizeof(_Float16));
    _Float16*  W2rT  = (_Float16*)alloc((size_t)F * F * sizeof(_Float16));
    _Float16*  Wc1T  = (_Float16*)alloc((size_t)F * KPAD_C * sizeof(_Float16));

    hipMemsetAsync(agg, 0, (size_t)N * F * sizeof(float), stream);
    hipMemsetAsync(cnt, 0, (size_t)N * sizeof(float), stream);

    const int TPB = 256;
    int nF = N * F;

    // weight prep (transposed, f16, zero-padded K for the classifier)
    gnn_w_transpose_f16<<<(F * F + TPB - 1) / TPB, TPB, 0, stream>>>(W1l, W1lT, F, F, F);
    gnn_w_transpose_f16<<<(F * F + TPB - 1) / TPB, TPB, 0, stream>>>(W1r, W1rT, F, F, F);
    gnn_w_transpose_f16<<<(F * F + TPB - 1) / TPB, TPB, 0, stream>>>(W2l, W2lT, F, F, F);
    gnn_w_transpose_f16<<<(F * F + TPB - 1) / TPB, TPB, 0, stream>>>(W2r, W2rT, F, F, F);
    gnn_w_transpose_f16<<<(F * KPAD_C + TPB - 1) / TPB, TPB, 0, stream>>>(Wc1, Wc1T, 2 * F + 3, F, KPAD_C);
    gnn_f32_to_f16<<<(nF + TPB - 1) / TPB, TPB, 0, stream>>>(x, xh, nF);

    // layer 1: scatter-mean + fused dual-GEMM + ReLU
    int scatterBlocks = (int)(((size_t)E * 32 + TPB - 1) / TPB);
    gnn_scatter_add<<<scatterBlocks, TPB, 0, stream>>>(x, src, dstI, agg, cnt, E);
    gnn_mean_to_f16<<<(nF + TPB - 1) / TPB, TPB, 0, stream>>>(agg, cnt, meanh, nF);
    int waves = (N + 15) / 16;
    int gemmBlocks = (waves + 3) / 4;              // 4 waves / block (128 threads)
    gnn_node_gemm<<<gemmBlocks, 128, 0, stream>>>(meanh, xh, W1lT, W1rT, b1, h1f, h1h, N, 1);

    // layer 2 (no ReLU on output; degree counts reused)
    hipMemsetAsync(agg, 0, (size_t)N * F * sizeof(float), stream);
    gnn_scatter_add<<<scatterBlocks, TPB, 0, stream>>>(h1f, src, dstI, agg, nullptr, E);
    gnn_mean_to_f16<<<(nF + TPB - 1) / TPB, TPB, 0, stream>>>(agg, cnt, meanh, nF);
    gnn_node_gemm<<<gemmBlocks, 128, 0, stream>>>(meanh, h1h, W2lT, W2rT, b2, nullptr, h2h, N, 0);

    // fused edge classifier
    int tiles = (E + 15) / 16;
    int edgeBlocks = (tiles + 7) / 8;              // 8 waves / block (256 threads)
    gnn_edge_mlp<<<edgeBlocks, TPB, 0, stream>>>(h2h, src, dstI, ea, Wc1T, bc1, Wc2, bc2, out, E);
}
